// AttentionWithRoPE_61701500175232
// MI455X (gfx1250) — matrix operand order repaired and checked by reading, not compiled, not run on hardware
//
#include <hip/hip_runtime.h>

// ---------------------------------------------------------------------------
// Types for CDNA5 WMMA (wave32): V_WMMA_F32_16X16X32_BF16
// ---------------------------------------------------------------------------
typedef __attribute__((ext_vector_type(16))) __bf16 bf16x16;
typedef __attribute__((ext_vector_type(8)))  float  f32x8;

__device__ __forceinline__ f32x8 wmma_bf16(bf16x16 a, bf16x16 b, f32x8 c) {
    return __builtin_amdgcn_wmma_f32_16x16x32_bf16(false, a, false, b,
                                                   (short)0, c, false, false);
}

// round-to-nearest-even f32 -> bf16 (raw u16)
__device__ __forceinline__ unsigned short f2bf(float f) {
    unsigned u = __float_as_uint(f);
    u += 0x7FFFu + ((u >> 16) & 1u);
    return (unsigned short)(u >> 16);
}

__device__ __forceinline__ bf16x16 ld_frag2(const unsigned short* p0,
                                            const unsigned short* p1) {
    uint4 a = *(const uint4*)p0;
    uint4 b = *(const uint4*)p1;
    union { unsigned int u[8]; bf16x16 v; } f;
    f.u[0] = a.x; f.u[1] = a.y; f.u[2] = a.z; f.u[3] = a.w;
    f.u[4] = b.x; f.u[5] = b.y; f.u[6] = b.z; f.u[7] = b.w;
    return f.v;
}

// ---------------------------------------------------------------------------
// Async global->LDS copy (gfx1250 GLOBAL_LOAD_ASYNC_TO_LDS_B128, ASYNCcnt)
// Builtin signature (from hipcc diagnostic): param0 = v4i addrspace(1)*,
// param1 = v4i addrspace(3)*, then imm offset, imm cpol.
// ---------------------------------------------------------------------------
typedef int v4i __attribute__((vector_size(16)));
typedef __attribute__((address_space(1))) v4i gv4i;
typedef __attribute__((address_space(3))) v4i sv4i;

__device__ __forceinline__ void cp_async_b128(const void* g, void* l) {
#if __has_builtin(__builtin_amdgcn_global_load_async_to_lds_b128)
    gv4i* gp = (gv4i*)(unsigned long long)(uintptr_t)g;
    // generic LDS addr low 32 bits == LDS offset (aperture truncation)
    sv4i* lp = (sv4i*)(unsigned)(uintptr_t)l;
    __builtin_amdgcn_global_load_async_to_lds_b128(gp, lp, 0, 0);
#elif defined(__gfx1250__)
    asm volatile("global_load_async_to_lds_b128 %0, %1, off"
                 :: "v"((unsigned)(uintptr_t)l),
                    "v"((unsigned long long)(uintptr_t)g)
                 : "memory");
#else
    *(uint4*)l = *(const uint4*)g;
#endif
}

__device__ __forceinline__ void wait_async0() {
#if __has_builtin(__builtin_amdgcn_s_wait_asynccnt)
    __builtin_amdgcn_s_wait_asynccnt(0);
#elif defined(__gfx1250__)
    asm volatile("s_wait_asynccnt 0x0" ::: "memory");
#endif
}

// ---------------------------------------------------------------------------
// Problem constants
// ---------------------------------------------------------------------------
#define BB   4
#define NSEQ 1024
#define CC   1024
#define HH   16
#define HD   64

// ---------------------------------------------------------------------------
// f32 -> bf16 conversion pass (bandwidth bound, one-shot)
// ---------------------------------------------------------------------------
__global__ __launch_bounds__(256)
void cvt_bf16_kernel(const float* __restrict__ src,
                     unsigned short* __restrict__ dst, int n4) {
    int i = blockIdx.x * 256 + threadIdx.x;
    if (i < n4) {
        float4 v = *(const float4*)(src + (size_t)i * 4);
        unsigned a0 = f2bf(v.x), a1 = f2bf(v.y), a2 = f2bf(v.z), a3 = f2bf(v.w);
        *(uint2*)(dst + (size_t)i * 4) = make_uint2(a0 | (a1 << 16), a2 | (a3 << 16));
    }
}

// ---------------------------------------------------------------------------
// GEMM  Out[M][Nout] = A[M][K] @ W[Nout][K]^T + bias   (A, W already bf16)
// WG tile 128x64, K-tile 64 (2 WMMA k-steps), 8 waves, wave tile 32x32.
// Staging is a verbatim byte copy -> async global->LDS.
// ---------------------------------------------------------------------------
__global__ __launch_bounds__(256)
void gemm_bf16_kernel(const unsigned short* __restrict__ A,
                      const unsigned short* __restrict__ W,
                      const float* __restrict__ bias, float* __restrict__ Out,
                      int M, int Nout, int K) {
    __shared__ unsigned short As[128][64];  // [m][k]
    __shared__ unsigned short Ws[64][64];   // [n][k]

    const int tid  = threadIdx.x;
    const int lane = tid & 31;
    const int wave = tid >> 5;
    const int wm   = wave & 3;
    const int wn   = wave >> 2;
    const int mBase = blockIdx.y * 128;
    const int nBase = blockIdx.x * 64;

    f32x8 acc[2][2] = {};

    for (int k0 = 0; k0 < K; k0 += 64) {
        // stage A tile 128x64 bf16: 1024 x 16B granules
        for (int i = tid; i < 128 * 8; i += 256) {
            int r = i >> 3, c = (i & 7) * 8;
            cp_async_b128(A + (size_t)(mBase + r) * K + k0 + c, &As[r][c]);
        }
        // stage W tile 64x64 bf16: 512 x 16B granules
        for (int i = tid; i < 64 * 8; i += 256) {
            int r = i >> 3, c = (i & 7) * 8;
            cp_async_b128(W + (size_t)(nBase + r) * K + k0 + c, &Ws[r][c]);
        }
        wait_async0();
        __syncthreads();

#pragma unroll
        for (int ks = 0; ks < 2; ++ks) {
            bf16x16 af[2], bfr[2];
            int rr = lane & 15;
            int c0 = ks * 32 + ((lane < 16) ? 0 : 8);
            af[0] = ld_frag2(&As[wm * 32 + rr][c0],      &As[wm * 32 + rr][c0 + 16]);
            af[1] = ld_frag2(&As[wm * 32 + 16 + rr][c0], &As[wm * 32 + 16 + rr][c0 + 16]);
            int kk = ks * 32 + ((lane < 16) ? 0 : 8);
            bfr[0] = ld_frag2(&Ws[wn * 32 + rr][kk],      &Ws[wn * 32 + rr][kk + 16]);
            bfr[1] = ld_frag2(&Ws[wn * 32 + 16 + rr][kk], &Ws[wn * 32 + 16 + rr][kk + 16]);
#pragma unroll
            for (int mi = 0; mi < 2; ++mi)
#pragma unroll
                for (int ni = 0; ni < 2; ++ni)
                    acc[mi][ni] = wmma_bf16(af[mi], bfr[ni], acc[mi][ni]);
        }
        __syncthreads();
    }

    const int hi = lane >> 4;
    const int cl = lane & 15;
#pragma unroll
    for (int mi = 0; mi < 2; ++mi)
#pragma unroll
        for (int ni = 0; ni < 2; ++ni) {
            int ncol = nBase + wn * 32 + ni * 16 + cl;
            float bv = bias[ncol];
#pragma unroll
            for (int r = 0; r < 8; ++r) {
                int mrow = mBase + wm * 32 + mi * 16 + r + 8 * hi;
                Out[(size_t)mrow * Nout + ncol] = acc[mi][ni][r] + bv;
            }
        }
}

// ---------------------------------------------------------------------------
// RoPE + scale + pack. Q,K -> [bh][n][hd] bf16;  V -> TRANSPOSED [bh][hd][n]
// so attention can stage V with verbatim async copies.
// Both hd^-0.5 scales folded into q (factor 1/64).
// ---------------------------------------------------------------------------
__global__ __launch_bounds__(256)
void rope_pack_kernel(const float* __restrict__ qkv,
                      unsigned short* __restrict__ Qp,
                      unsigned short* __restrict__ Kp,
                      unsigned short* __restrict__ Vt) {
    int idx = blockIdx.x * 256 + threadIdx.x;  // B*N*H*32 threads exactly
    int p = idx & 31; int t = idx >> 5;
    int h = t & 15;   t >>= 4;
    int n = t & 1023; t >>= 10;
    int b = t;

    const size_t base = (size_t)(b * NSEQ + n) * (3 * CC);
    int d = 2 * p;
    float q0 = qkv[base + h * HD + d],          q1 = qkv[base + h * HD + d + 1];
    float k0 = qkv[base + CC + h * HD + d],     k1 = qkv[base + CC + h * HD + d + 1];
    float v0 = qkv[base + 2 * CC + h * HD + d], v1 = qkv[base + 2 * CC + h * HD + d + 1];

    float inv = __expf(-9.2103403719761836f * ((float)d * (1.0f / (float)HD)));
    float fr  = (float)n * inv;
    float c = __cosf(fr), s = __sinf(fr);

    const float sc = 1.0f / 64.0f;  // (hd^-0.5)^2
    float qr0 = (q0 * c - q1 * s) * sc;
    float qr1 = (q1 * c + q0 * s) * sc;
    float kr0 = (k0 * c - k1 * s);
    float kr1 = (k1 * c + k0 * s);

    int bh = b * HH + h;
    size_t o = ((size_t)bh * NSEQ + n) * HD + d;
    Qp[o] = f2bf(qr0); Qp[o + 1] = f2bf(qr1);
    Kp[o] = f2bf(kr0); Kp[o + 1] = f2bf(kr1);
    size_t ov = ((size_t)bh * HD + d) * NSEQ + n;   // transposed
    Vt[ov] = f2bf(v0); Vt[ov + NSEQ] = f2bf(v1);
}

// ---------------------------------------------------------------------------
// Flash attention: WG = (b,h) x 128 q-rows, 8 waves x 16 rows.
// K chunk [128][64] and V chunk [64][128] staged with async b128 copies.
// Output written directly as bf16 [B][N][C].
// ---------------------------------------------------------------------------
__global__ __launch_bounds__(256)
void attn_kernel(const unsigned short* __restrict__ Q,
                 const unsigned short* __restrict__ Kg,
                 const unsigned short* __restrict__ Vt,
                 unsigned short* __restrict__ Out) {
    __shared__ unsigned short Ks[128][64];     // [key][hd]
    __shared__ unsigned short Vs[64][128];     // [hd][key]
    __shared__ unsigned short Ps[8][16 * 32];  // per-wave P scratch

    const int tid = threadIdx.x, lane = tid & 31, wave = tid >> 5;
    const int bh = blockIdx.y;
    const int b  = bh >> 4, h = bh & 15;
    const int qBase = blockIdx.x * 128 + wave * 16;

    const unsigned short* qrow = Q + ((size_t)bh * NSEQ + qBase) * HD;
    bf16x16 qf[2];
    {
        int rr = lane & 15;
        int c0 = (lane < 16) ? 0 : 8;
        qf[0] = ld_frag2(qrow + rr * HD + c0,      qrow + rr * HD + c0 + 16);
        qf[1] = ld_frag2(qrow + rr * HD + 32 + c0, qrow + rr * HD + 32 + c0 + 16);
    }

    f32x8 O[4] = {};
    float Mrow[8], Lrow[8];
#pragma unroll
    for (int r = 0; r < 8; ++r) { Mrow[r] = -3.0e38f; Lrow[r] = 0.0f; }

    for (int ks = 0; ks < NSEQ; ks += 128) {
        // stage K chunk: 128 rows x 4 granules/thread
        for (int i = tid; i < 128 * 8; i += 256) {
            int r = i >> 3, c = (i & 7) * 8;
            cp_async_b128(Kg + ((size_t)bh * NSEQ + ks + r) * HD + c, &Ks[r][c]);
        }
        // stage V chunk (already transposed in global): 64 rows x 16 granules
        for (int i = tid; i < 64 * 16; i += 256) {
            int r = i >> 4, c = (i & 15) * 8;
            cp_async_b128(Vt + ((size_t)bh * HD + r) * NSEQ + ks + c, &Vs[r][c]);
        }
        wait_async0();
        __syncthreads();

        for (int so = 0; so < 128; so += 32) {
            // ---- S = Q x K^T over 32 keys ----
            f32x8 S0 = {}, S1 = {};
            {
                int key0 = so + (lane & 15);
                int key1 = key0 + 16;
                int k0   = (lane < 16) ? 0 : 8;
                bf16x16 bk;
                bk = ld_frag2(&Ks[key0][k0],      &Ks[key0][k0 + 16]);      S0 = wmma_bf16(qf[0], bk, S0);
                bk = ld_frag2(&Ks[key0][32 + k0], &Ks[key0][32 + k0 + 16]); S0 = wmma_bf16(qf[1], bk, S0);
                bk = ld_frag2(&Ks[key1][k0],      &Ks[key1][k0 + 16]);      S1 = wmma_bf16(qf[0], bk, S1);
                bk = ld_frag2(&Ks[key1][32 + k0], &Ks[key1][32 + k0 + 16]); S1 = wmma_bf16(qf[1], bk, S1);
            }

            // ---- online softmax ----
            float alpha[8];
#pragma unroll
            for (int r = 0; r < 8; ++r) {
                float v = fmaxf(S0[r], S1[r]);
                v = fmaxf(v, __shfl_xor(v, 1, 32));
                v = fmaxf(v, __shfl_xor(v, 2, 32));
                v = fmaxf(v, __shfl_xor(v, 4, 32));
                v = fmaxf(v, __shfl_xor(v, 8, 32));
                float mn = fmaxf(Mrow[r], v);
                alpha[r] = __expf(Mrow[r] - mn);
                Mrow[r] = mn;
            }
            float P0[8], P1[8];
#pragma unroll
            for (int r = 0; r < 8; ++r) {
                P0[r] = __expf(S0[r] - Mrow[r]);
                P1[r] = __expf(S1[r] - Mrow[r]);
                float s = P0[r] + P1[r];
                s += __shfl_xor(s, 1, 32);
                s += __shfl_xor(s, 2, 32);
                s += __shfl_xor(s, 4, 32);
                s += __shfl_xor(s, 8, 32);
                Lrow[r] = Lrow[r] * alpha[r] + s;
            }
#pragma unroll
            for (int nb = 0; nb < 4; ++nb)
#pragma unroll
                for (int r = 0; r < 8; ++r) O[nb][r] *= alpha[r];

            // ---- P (C-layout) -> bf16 A-frag via per-wave LDS ----
            {
                int hi2 = lane >> 4, cc = lane & 15;
                unsigned short* ps = &Ps[wave][0];
#pragma unroll
                for (int r = 0; r < 8; ++r) {
                    int row = r + 8 * hi2;
                    ps[row * 32 + cc]      = f2bf(P0[r]);
                    ps[row * 32 + 16 + cc] = f2bf(P1[r]);
                }
            }
            asm volatile("s_wait_dscnt 0x0" ::: "memory");  // DS in-order per wave
            bf16x16 pf;
            {
                int rr = lane & 15;
                int c0 = (lane < 16) ? 0 : 8;
                const unsigned short* ps = &Ps[wave][0];
                pf = ld_frag2(ps + rr * 32 + c0, ps + rr * 32 + c0 + 16);
            }

            // ---- O += P x V ----
            {
                int hd0 = lane & 15;
                int k0  = so + ((lane < 16) ? 0 : 8);
#pragma unroll
                for (int nb = 0; nb < 4; ++nb) {
                    bf16x16 bv = ld_frag2(&Vs[nb * 16 + hd0][k0],
                                          &Vs[nb * 16 + hd0][k0 + 16]);
                    O[nb] = wmma_bf16(pf, bv, O[nb]);
                }
            }
        }
        __syncthreads();
    }

    // epilogue: normalize, write bf16 [B][N][C]
    {
        int hi2 = lane >> 4, cc = lane & 15;
#pragma unroll
        for (int r = 0; r < 8; ++r) {
            float invl = 1.0f / Lrow[r];
            int m = qBase + r + 8 * hi2;
            unsigned short* orow = Out + ((size_t)b * NSEQ + m) * CC + h * HD;
#pragma unroll
            for (int nb = 0; nb < 4; ++nb)
                orow[nb * 16 + cc] = f2bf(O[nb][r] * invl);
        }
    }
}

// ---------------------------------------------------------------------------
// Workspace layout (bytes):
//   [ 0,  8MB)  x bf16        [4096][1024]
//   [ 8, 14MB)  qkv_w bf16    [3072][1024]
//   [14, 16MB)  proj_w bf16   [1024][1024]
//   [16, 64MB)  qkv f32       [4096][3072]
//   [64, 72MB)  Q bf16        [64][1024][64]
//   [72, 80MB)  K bf16        [64][1024][64]
//   [80, 88MB)  V^T bf16      [64][64][1024]
//   [88, 96MB)  attn bf16     [4096][1024]
// ---------------------------------------------------------------------------
extern "C" void kernel_launch(void* const* d_in, const int* in_sizes, int n_in,
                              void* d_out, int out_size, void* d_ws, size_t ws_size,
                              hipStream_t stream) {
    const float* x      = (const float*)d_in[0];
    const float* qkv_w  = (const float*)d_in[1];
    const float* qkv_b  = (const float*)d_in[2];
    const float* proj_w = (const float*)d_in[3];
    const float* proj_b = (const float*)d_in[4];
    float* out = (float*)d_out;

    char* ws = (char*)d_ws;
    const size_t MB = 1024ull * 1024ull;
    unsigned short* xb    = (unsigned short*)(ws);
    unsigned short* qwb   = (unsigned short*)(ws + 8 * MB);
    unsigned short* pwb   = (unsigned short*)(ws + 14 * MB);
    float*          qkv   = (float*)(ws + 16 * MB);
    unsigned short* Qp    = (unsigned short*)(ws + 64 * MB);
    unsigned short* Kp    = (unsigned short*)(ws + 72 * MB);
    unsigned short* Vt    = (unsigned short*)(ws + 80 * MB);
    unsigned short* attnb = (unsigned short*)(ws + 88 * MB);

    dim3 blk(256);

    // bf16 pre-conversion (keeps GEMM staging VALU-free)
    {
        int n4 = (BB * NSEQ * CC) / 4;
        cvt_bf16_kernel<<<dim3((n4 + 255) / 256), blk, 0, stream>>>(x, xb, n4);
        n4 = (3 * CC * CC) / 4;
        cvt_bf16_kernel<<<dim3((n4 + 255) / 256), blk, 0, stream>>>(qkv_w, qwb, n4);
        n4 = (CC * CC) / 4;
        cvt_bf16_kernel<<<dim3((n4 + 255) / 256), blk, 0, stream>>>(proj_w, pwb, n4);
    }

    // QKV GEMM: M=4096, Nout=3072, K=1024
    gemm_bf16_kernel<<<dim3(3072 / 64, 4096 / 128), blk, 0, stream>>>(
        xb, qwb, qkv_b, qkv, BB * NSEQ, 3 * CC, CC);

    // RoPE + pack
    rope_pack_kernel<<<dim3((BB * NSEQ * HH * 32) / 256), blk, 0, stream>>>(
        qkv, Qp, Kp, Vt);

    // Attention
    attn_kernel<<<dim3(NSEQ / 128, BB * HH), blk, 0, stream>>>(Qp, Kp, Vt, attnb);

    // Projection GEMM: M=4096, Nout=1024, K=1024
    gemm_bf16_kernel<<<dim3(1024 / 64, 4096 / 128), blk, 0, stream>>>(
        attnb, pwb, proj_b, out, BB * NSEQ, CC, CC);
}
